// MultiHeadAttention_81509889343615
// MI455X (gfx1250) — matrix operand order repaired and checked by reading, not compiled, not run on hardware
//
#include <hip/hip_runtime.h>
#include <hip/hip_bf16.h>
#include <hip/hip_fp16.h>

#define BDIM   2
#define LSEQ   2048
#define DMODEL 1024
#define NHEAD  16
#define HDIM   64

typedef _Float16 h8  __attribute__((ext_vector_type(8)));
typedef _Float16 h16 __attribute__((ext_vector_type(16)));
typedef float    f8  __attribute__((ext_vector_type(8)));
typedef float    f4  __attribute__((ext_vector_type(4)));
typedef unsigned int u4 __attribute__((ext_vector_type(4)));

// ---- WMMA fragment loaders (per CDNA5 ISA 7.12.2 layouts, wave32) -----------
// A (16x32 f16): lane l holds row M=l%16; element e: K = (e<8 ? e : e+8) + 8*(l/16)
//   -> two contiguous 8-half (16B) chunks at k0+8*half and k0+16+8*half.
static __device__ __forceinline__ h16 ldsA(const _Float16* base, int row0, int k0,
                                           int ld, int lane) {
  const int r = lane & 15, hf = lane >> 4;
  const _Float16* p = base + (row0 + r) * ld + k0 + 8 * hf;
  h8 lo = *(const h8*)(p);
  h8 hi = *(const h8*)(p + 16);
  return __builtin_shufflevector(lo, hi, 0,1,2,3,4,5,6,7,8,9,10,11,12,13,14,15);
}
// B (32x16 f16): lane l holds column N=l%16; K = (e<8 ? e : e+8) + 8*(l/16)
//   -> two contiguous 8-half (16B) chunks. base is stored [N][K] row-major.
static __device__ __forceinline__ h16 ldsB(const _Float16* base, int col0, int k0,
                                           int ld, int lane) {
  const int n = lane & 15, hf = lane >> 4;
  const _Float16* p = base + (col0 + n) * ld + k0 + 8 * hf;
  h8 lo = *(const h8*)(p);
  h8 hi = *(const h8*)(p + 16);
  return __builtin_shufflevector(lo, hi, 0,1,2,3,4,5,6,7,8,9,10,11,12,13,14,15);
}
static __device__ __forceinline__ f8 wmma16(h16 a, h16 b, f8 c) {
  return __builtin_amdgcn_wmma_f32_16x16x32_f16(false, a, false, b, (short)0, c,
                                                false, false);
}

// ============================================================================
// Kernel 1: QKV projections. out = in @ W^T, f16 results.
//   z=0: Q -> Qh (B,H,L,HD)   z=1: K -> Kh (B,H,L,HD)   z=2: V -> Vt (B,H,HD,L)
// N-tile of 64 == exactly one head (HD=64).
// ============================================================================
__global__ __launch_bounds__(256) void qkv_proj_kernel(
    const float* __restrict__ qin, const float* __restrict__ kin,
    const float* __restrict__ vin, const float* __restrict__ wq,
    const float* __restrict__ wk, const float* __restrict__ wv,
    _Float16* __restrict__ Qh, _Float16* __restrict__ Kh,
    _Float16* __restrict__ Vt) {
  __shared__ __align__(32) _Float16 As[64 * 32];
  __shared__ __align__(32) _Float16 Bs[64 * 32];
  __shared__ __align__(32) _Float16 Cs[64 * 64];

  const int tid = threadIdx.x;
  const int lane = tid & 31, w = tid >> 5;
  const int m0 = blockIdx.x * 64;  // row in flattened (B*L)
  const int h  = blockIdx.y;       // head index
  const int z  = blockIdx.z;
  const float* src = (z == 0) ? qin : (z == 1) ? kin : vin;
  const float* wgt = (z == 0) ? wq  : (z == 1) ? wk  : wv;

  const int lrow = tid >> 2;        // 0..63
  const int lcol = (tid & 3) * 8;   // 0,8,16,24
  const int rsub = 16 * (w & 3);    // wave M-subtile
  const int nsub = 32 * (w >> 2);   // wave N-subtile (2x 16 cols)

  f8 acc0 = {0, 0, 0, 0, 0, 0, 0, 0};
  f8 acc1 = {0, 0, 0, 0, 0, 0, 0, 0};

  for (int kk = 0; kk < DMODEL; kk += 32) {
    __syncthreads();
    {
      const f4* pa = (const f4*)(src + (size_t)(m0 + lrow) * DMODEL + kk + lcol);
      f4 a0 = pa[0], a1 = pa[1];
      h8 ha = {(_Float16)a0.x, (_Float16)a0.y, (_Float16)a0.z, (_Float16)a0.w,
               (_Float16)a1.x, (_Float16)a1.y, (_Float16)a1.z, (_Float16)a1.w};
      *(h8*)&As[lrow * 32 + lcol] = ha;
      const f4* pb =
          (const f4*)(wgt + (size_t)(h * 64 + lrow) * DMODEL + kk + lcol);
      f4 b0 = pb[0], b1 = pb[1];
      h8 hb = {(_Float16)b0.x, (_Float16)b0.y, (_Float16)b0.z, (_Float16)b0.w,
               (_Float16)b1.x, (_Float16)b1.y, (_Float16)b1.z, (_Float16)b1.w};
      *(h8*)&Bs[lrow * 32 + lcol] = hb;
      if (kk + 32 < DMODEL)
        __builtin_prefetch(src + (size_t)(m0 + lrow) * DMODEL + kk + 32 + lcol, 0, 3);
    }
    __syncthreads();
    h16 a = ldsA(As, rsub, 0, 32, lane);
    acc0 = wmma16(a, ldsB(Bs, nsub + 0, 0, 32, lane), acc0);
    acc1 = wmma16(a, ldsB(Bs, nsub + 16, 0, 32, lane), acc1);
  }

  const int n = lane & 15, hf = lane >> 4;
  const int b  = m0 >> 11;      // m0 / LSEQ
  const int l0 = m0 & (LSEQ - 1);
  if (z < 2) {
    _Float16* dst = (z == 0) ? Qh : Kh;
#pragma unroll
    for (int vv = 0; vv < 8; ++vv) {
      const int row = rsub + vv + 8 * hf;
      const size_t base =
          (((size_t)(b * NHEAD + h)) * LSEQ + (l0 + row)) * HDIM;
      dst[base + nsub + n]      = (_Float16)acc0[vv];
      dst[base + nsub + 16 + n] = (_Float16)acc1[vv];
    }
  } else {
    // bounce through LDS, store transposed: Vt[(b*H+h)*HD + hd][l]
#pragma unroll
    for (int vv = 0; vv < 8; ++vv) {
      const int row = rsub + vv + 8 * hf;
      Cs[row * 64 + nsub + n]      = (_Float16)acc0[vv];
      Cs[row * 64 + nsub + 16 + n] = (_Float16)acc1[vv];
    }
    __syncthreads();
    const int hd = tid >> 2;
    const int mc = (tid & 3) * 16;
    union { _Float16 hh[16]; u4 u[2]; } pack;
#pragma unroll
    for (int i = 0; i < 16; ++i) pack.hh[i] = Cs[(mc + i) * 64 + hd];
    _Float16* dst =
        Vt + (((size_t)(b * NHEAD + h)) * HDIM + hd) * LSEQ + l0 + mc;
    ((u4*)dst)[0] = pack.u[0];
    ((u4*)dst)[1] = pack.u[1];
  }
}

// ============================================================================
// Kernel 2: causal attention, two-sweep (stats, then emit P + accumulate P*V).
// One workgroup per (q-block of 64 rows, b*H+h). 8 waves; wave w owns
// S rows 16*(w>>1) x cols 32*(w&1).
// ============================================================================
__global__ __launch_bounds__(256) void attn_kernel(
    const _Float16* __restrict__ Qh, const _Float16* __restrict__ Kh,
    const _Float16* __restrict__ Vt, float* __restrict__ attn,
    _Float16* __restrict__ Ctx) {
  __shared__ __align__(32) _Float16 Qs[64 * 64];
  __shared__ __align__(32) _Float16 Ks[64 * 64];
  __shared__ __align__(32) _Float16 Vts[64 * 64];  // [d][key]
  __shared__ __align__(32) _Float16 Ps[64 * 64];
  __shared__ float Ssc[64 * 64];
  __shared__ float rowm[64], rowl[64];
  __shared__ float pm[64][4], pl[64][4];

  const float NEG_INF = -__builtin_inff();
  const int tid = threadIdx.x, lane = tid & 31, w = tid >> 5;
  const int n = lane & 15, hf = lane >> 4;
  const int qt = blockIdx.x, bh = blockIdx.y;
  const int q0 = qt * 64;
  const int rsub = 16 * (w >> 1);
  const int csub = 32 * (w & 1);
  const size_t qkBase = (size_t)bh * LSEQ * HDIM;

  const int lrow = tid >> 2;        // 0..63
  const int lc16 = (tid & 3) * 16;  // 0,16,32,48

  *(u4*)&Qs[lrow * 64 + lc16] =
      *(const u4*)(Qh + qkBase + (size_t)(q0 + lrow) * HDIM + lc16);
  if (tid < 64) { rowm[tid] = NEG_INF; rowl[tid] = 0.f; }

  const int jq = qt;

  // ---------------- Sweep 1: per-row max / expsum (online) ----------------
  for (int j = 0; j <= jq; ++j) {
    const int j0 = j * 64;
    __syncthreads();
    *(u4*)&Ks[lrow * 64 + lc16] =
        *(const u4*)(Kh + qkBase + (size_t)(j0 + lrow) * HDIM + lc16);
    __syncthreads();
#pragma unroll
    for (int c = 0; c < 2; ++c) {
      f8 acc = {0, 0, 0, 0, 0, 0, 0, 0};
#pragma unroll
      for (int k0 = 0; k0 < HDIM; k0 += 32)
        acc = wmma16(ldsA(Qs, rsub, k0, 64, lane),
                     ldsB(Ks, csub + 16 * c, k0, 64, lane), acc);
#pragma unroll
      for (int vv = 0; vv < 8; ++vv)
        Ssc[(rsub + vv + 8 * hf) * 64 + csub + 16 * c + n] = acc[vv];
    }
    __syncthreads();
    {  // partial stats: 4 threads per row, 16 cols each
      const int r = lrow, rg = q0 + r;
      float sv[16];
      float ml = NEG_INF;
#pragma unroll
      for (int i = 0; i < 16; ++i) {
        const int c = lc16 + i, cg = j0 + c;
        float s = 0.125f * Ssc[r * 64 + c];
        const bool mask = (cg > rg) || (s == 0.0f);  // tril==0 quirk
        s = mask ? NEG_INF : s;
        sv[i] = s;
        ml = fmaxf(ml, s);
      }
      float sl = 0.f;
      if (ml != NEG_INF) {
#pragma unroll
        for (int i = 0; i < 16; ++i) sl += __expf(sv[i] - ml);
      }
      pm[r][tid & 3] = ml;
      pl[r][tid & 3] = sl;
    }
    __syncthreads();
    if (tid < 64) {  // combine partials with running stats
      const float m_old = rowm[tid], l_old = rowl[tid];
      float mt = m_old;
#pragma unroll
      for (int i = 0; i < 4; ++i) mt = fmaxf(mt, pm[tid][i]);
      if (mt != NEG_INF) {
        float ln = (l_old > 0.f) ? l_old * __expf(m_old - mt) : 0.f;
#pragma unroll
        for (int i = 0; i < 4; ++i)
          if (pl[tid][i] > 0.f) ln += pl[tid][i] * __expf(pm[tid][i] - mt);
        rowm[tid] = mt;
        rowl[tid] = ln;
      }
    }
  }

  // ---------------- Sweep 2: recompute S, emit P, accumulate O = P*V --------
  f8 accO0 = {0, 0, 0, 0, 0, 0, 0, 0};
  f8 accO1 = {0, 0, 0, 0, 0, 0, 0, 0};
  for (int j = 0; j <= jq; ++j) {
    const int j0 = j * 64;
    __syncthreads();
    *(u4*)&Ks[lrow * 64 + lc16] =
        *(const u4*)(Kh + qkBase + (size_t)(j0 + lrow) * HDIM + lc16);
    *(u4*)&Vts[lrow * 64 + lc16] =
        *(const u4*)(Vt + qkBase + (size_t)lrow * LSEQ + j0 + lc16);
    __syncthreads();
#pragma unroll
    for (int c = 0; c < 2; ++c) {
      f8 acc = {0, 0, 0, 0, 0, 0, 0, 0};
#pragma unroll
      for (int k0 = 0; k0 < HDIM; k0 += 32)
        acc = wmma16(ldsA(Qs, rsub, k0, 64, lane),
                     ldsB(Ks, csub + 16 * c, k0, 64, lane), acc);
#pragma unroll
      for (int vv = 0; vv < 8; ++vv)
        Ssc[(rsub + vv + 8 * hf) * 64 + csub + 16 * c + n] = acc[vv];
    }
    __syncthreads();
    {  // P = softmax row-slice; write fp32 attention (streaming) + f16 Ps
      const int r = lrow, rg = q0 + r;
      const float mR = rowm[r], lR = rowl[r];
      const float invl = (lR > 0.f) ? 1.f / lR : 0.f;
      float pv[16];
#pragma unroll
      for (int i = 0; i < 16; ++i) {
        const int c = lc16 + i, cg = j0 + c;
        const float s = 0.125f * Ssc[r * 64 + c];
        const bool mask = (cg > rg) || (s == 0.0f);
        const float p = mask ? 0.f : __expf(s - mR) * invl;
        pv[i] = p;
        Ps[r * 64 + c] = (_Float16)p;
      }
      float* dst = attn + ((size_t)bh * LSEQ + (q0 + r)) * LSEQ + j0 + lc16;
#pragma unroll
      for (int i = 0; i < 4; ++i) {
        f4 o4 = {pv[4 * i], pv[4 * i + 1], pv[4 * i + 2], pv[4 * i + 3]};
        __builtin_nontemporal_store(o4, (f4*)dst + i);
      }
    }
    __syncthreads();
#pragma unroll
    for (int k0 = 0; k0 < 64; k0 += 32) {
      h16 a = ldsA(Ps, rsub, k0, 64, lane);
      accO0 = wmma16(a, ldsB(Vts, csub, k0, 64, lane), accO0);
      accO1 = wmma16(a, ldsB(Vts, csub + 16, k0, 64, lane), accO1);
    }
  }

  // write context (f16) into ws Ctx (B, L, D) with head interleave
  const int b0 = bh >> 4, hh = bh & 15;
#pragma unroll
  for (int vv = 0; vv < 8; ++vv) {
    const int row = rsub + vv + 8 * hf;
    const size_t base =
        ((size_t)b0 * LSEQ + q0 + row) * DMODEL + hh * HDIM;
    Ctx[base + csub + n]      = (_Float16)accO0[vv];
    Ctx[base + csub + 16 + n] = (_Float16)accO1[vv];
  }

  // zero-fill the strictly-upper key tiles of the attention output
  const f4 z4 = {0.f, 0.f, 0.f, 0.f};
  for (int j = jq + 1; j < LSEQ / 64; ++j) {
    float* dst =
        attn + ((size_t)bh * LSEQ + (q0 + lrow)) * LSEQ + j * 64 + lc16;
#pragma unroll
    for (int i = 0; i < 4; ++i)
      __builtin_nontemporal_store(z4, (f4*)dst + i);
  }
}

// ============================================================================
// Kernel 3: output projection. out = Ctx(f16) @ Wo^T, fp32 result.
// ============================================================================
__global__ __launch_bounds__(256) void out_proj_kernel(
    const _Float16* __restrict__ Ctx, const float* __restrict__ wo,
    float* __restrict__ out) {
  __shared__ __align__(32) _Float16 As[64 * 32];
  __shared__ __align__(32) _Float16 Bs[64 * 32];

  const int tid = threadIdx.x, lane = tid & 31, w = tid >> 5;
  const int m0 = blockIdx.x * 64, n0 = blockIdx.y * 64;
  const int lrow = tid >> 2, lcol = (tid & 3) * 8;
  const int rsub = 16 * (w & 3), nsub = 32 * (w >> 2);

  f8 acc0 = {0, 0, 0, 0, 0, 0, 0, 0};
  f8 acc1 = {0, 0, 0, 0, 0, 0, 0, 0};

  for (int kk = 0; kk < DMODEL; kk += 32) {
    __syncthreads();
    *(h8*)&As[lrow * 32 + lcol] =
        *(const h8*)(Ctx + (size_t)(m0 + lrow) * DMODEL + kk + lcol);
    {
      const f4* pb =
          (const f4*)(wo + (size_t)(n0 + lrow) * DMODEL + kk + lcol);
      f4 b0 = pb[0], b1 = pb[1];
      h8 hb = {(_Float16)b0.x, (_Float16)b0.y, (_Float16)b0.z, (_Float16)b0.w,
               (_Float16)b1.x, (_Float16)b1.y, (_Float16)b1.z, (_Float16)b1.w};
      *(h8*)&Bs[lrow * 32 + lcol] = hb;
    }
    __syncthreads();
    h16 a = ldsA(As, rsub, 0, 32, lane);
    acc0 = wmma16(a, ldsB(Bs, nsub + 0, 0, 32, lane), acc0);
    acc1 = wmma16(a, ldsB(Bs, nsub + 16, 0, 32, lane), acc1);
  }

  const int n = lane & 15, hf = lane >> 4;
#pragma unroll
  for (int vv = 0; vv < 8; ++vv) {
    const int row = m0 + rsub + vv + 8 * hf;
    out[(size_t)row * DMODEL + n0 + nsub + n]      = acc0[vv];
    out[(size_t)row * DMODEL + n0 + nsub + 16 + n] = acc1[vv];
  }
}

// ============================================================================
extern "C" void kernel_launch(void* const* d_in, const int* in_sizes, int n_in,
                              void* d_out, int out_size, void* d_ws,
                              size_t ws_size, hipStream_t stream) {
  const float* qin = (const float*)d_in[0];
  const float* kin = (const float*)d_in[1];
  const float* vin = (const float*)d_in[2];
  const float* wq  = (const float*)d_in[3];
  const float* wk  = (const float*)d_in[4];
  const float* wv  = (const float*)d_in[5];
  const float* wo  = (const float*)d_in[6];

  // workspace carve-up (f16): Qh/Kh (B,H,L,HD), Vt (B,H,HD,L), Ctx (B,L,D)
  const size_t elems = (size_t)BDIM * NHEAD * LSEQ * HDIM;  // 4,194,304
  char* ws = (char*)d_ws;
  _Float16* Qh  = (_Float16*)(ws);
  _Float16* Kh  = (_Float16*)(ws + 2 * elems);
  _Float16* Vt  = (_Float16*)(ws + 4 * elems);
  _Float16* Ctx = (_Float16*)(ws + 6 * elems);

  float* attn    = (float*)d_out;                              // (B,H,L,L)
  float* ctx_out = attn + (size_t)BDIM * NHEAD * LSEQ * LSEQ;  // (B,L,D)

  qkv_proj_kernel<<<dim3(BDIM * LSEQ / 64, NHEAD, 3), 256, 0, stream>>>(
      qin, kin, vin, wq, wk, wv, Qh, Kh, Vt);
  attn_kernel<<<dim3(LSEQ / 64, BDIM * NHEAD), 256, 0, stream>>>(
      Qh, Kh, Vt, attn, Ctx);
  out_proj_kernel<<<dim3(BDIM * LSEQ / 64, DMODEL / 64), 256, 0, stream>>>(
      Ctx, wo, ctx_out);
}
